// SpGAT_60696477827759
// MI455X (gfx1250) — matrix operand order repaired and checked by reading, not compiled, not run on hardware
//
#include <hip/hip_runtime.h>
#include <hip/hip_bf16.h>

typedef __attribute__((ext_vector_type(16))) _Float16 v16h;
typedef __attribute__((ext_vector_type(8)))  _Float16 v8h;
typedef __attribute__((ext_vector_type(8)))  float    v8f;

#define GAT_ALPHA 0.2f
#define GAT_LN_EPS 1e-5f

static constexpr int NN = 50000;   // nodes
static constexpr int EE = 800000;  // edges
static constexpr int FF = 256;     // input/output feature dim
static constexpr int DD = 64;      // head dim
static constexpr int HH = 8;       // heads
static constexpr int DH = DD * HH; // 512

// ---------------------------------------------------------------- utilities
__global__ void k_f32_to_f16(const float* __restrict__ src,
                             _Float16* __restrict__ dst, int n) {
  int i = blockIdx.x * blockDim.x + threadIdx.x;
  if (i < n) dst[i] = (_Float16)src[i];
}

__global__ void k_zero_f32(float* __restrict__ p, int n) {
  int i = blockIdx.x * blockDim.x + threadIdx.x;
  if (i < n) p[i] = 0.0f;
}

// ---------------------------------------------------------------- WMMA GEMM
// C[M,Nc] = A[M,K] (f16, row-major) x B[Nc,K] (f16, row-major == W, giving
// C = A @ W^T) + bias[Nc].  One wave computes a 16-row x 64-col slab
// (4 x v_wmma_f32_16x16x32_f16 accumulators, A fragment reused 4x).
__global__ void k_gemm_wmma(const _Float16* __restrict__ A,
                            const _Float16* __restrict__ B,
                            const float* __restrict__ bias,
                            float* __restrict__ C,
                            int Nc, int K, int tasks) {
  int wave = (int)((blockIdx.x * blockDim.x + threadIdx.x) >> 5);
  if (wave >= tasks) return;                 // uniform per-wave
  int groups = Nc >> 6;                      // 64-column groups
  int row_base = (wave / groups) << 4;
  int col_base = (wave % groups) << 6;
  int lane = threadIdx.x & 31;
  int l15  = lane & 15;
  int hi   = lane >> 4;                      // 0 for lanes 0-15, 1 for 16-31

  // A fragment base: row (row_base+l15); hi lanes shift K by +16 (contiguous)
  const _Float16* Ap = A + (size_t)(row_base + l15) * K + hi * 16;

  v8f acc0 = {}, acc1 = {}, acc2 = {}, acc3 = {};
  for (int k0 = 0; k0 < K; k0 += 32) {
    v8h a_lo = *(const v8h*)(Ap + k0);        // K = k0+{0..7}  (+16 if hi)
    v8h a_hi = *(const v8h*)(Ap + k0 + 8);    // K = k0+{8..15} (+16 if hi)
    v16h afrag = __builtin_shufflevector(a_lo, a_hi,
        0,1,2,3,4,5,6,7,8,9,10,11,12,13,14,15);
#pragma unroll
    for (int t = 0; t < 4; ++t) {
      // B fragment: column (col_base+t*16+l15) of C == row of W; 16
      // contiguous halfs starting at k0 (+16 for hi lanes).
      const _Float16* Bp =
          B + (size_t)(col_base + t * 16 + l15) * K + hi * 16 + k0;
      v16h bfrag = *(const v16h*)Bp;
      v8f& acc = (t == 0 ? acc0 : t == 1 ? acc1 : t == 2 ? acc2 : acc3);
      acc = __builtin_amdgcn_wmma_f32_16x16x32_f16(
          false, afrag, false, bfrag, (short)0, acc, false, false);
    }
  }

#pragma unroll
  for (int t = 0; t < 4; ++t) {
    v8f acc = (t == 0 ? acc0 : t == 1 ? acc1 : t == 2 ? acc2 : acc3);
    int col = col_base + t * 16 + l15;
    float bv = bias[col];
#pragma unroll
    for (int g = 0; g < 8; ++g) {
      int row = row_base + g + hi * 8;       // C/D layout, ISA 7.12.2
      C[(size_t)row * Nc + col] = acc[g] + bv;
    }
  }
}

// ---------------------------------------------------------------- layer 1
// Per-(node,head) attention logits: s_src = h . a[:D], s_dst = h . a[D:]
__global__ void k_scores1(const float* __restrict__ h,
                          const float* __restrict__ a,
                          float* __restrict__ ssrc,
                          float* __restrict__ sdst) {
  int wave = (int)((blockIdx.x * blockDim.x + threadIdx.x) >> 5);
  int n = wave >> 3, head = wave & 7;
  if (n >= NN) return;
  int lane = threadIdx.x & 31;
  float2 hv = ((const float2*)(h + (size_t)n * DH + head * DD))[lane];
  float2 a0 = ((const float2*)(a + head * 2 * DD))[lane];
  float2 a1 = ((const float2*)(a + head * 2 * DD + DD))[lane];
  float s0 = hv.x * a0.x + hv.y * a0.y;
  float s1 = hv.x * a1.x + hv.y * a1.y;
#pragma unroll
  for (int m = 16; m >= 1; m >>= 1) {
    s0 += __shfl_xor(s0, m, 32);
    s1 += __shfl_xor(s1, m, 32);
  }
  if (lane == 0) { ssrc[n * HH + head] = s0; sdst[n * HH + head] = s1; }
}

// Edge scatter: accum[src] += exp(-leakyrelu(s)) * h[dst], one wave per
// (edge, head), float2 per lane, native global_atomic_add_f32.
__global__ void k_scatter1(const float* __restrict__ h,
                           const int* __restrict__ ei,
                           const float* __restrict__ ssrc,
                           const float* __restrict__ sdst,
                           float* __restrict__ accum) {
  int wave = (int)((blockIdx.x * blockDim.x + threadIdx.x) >> 5);
  int e = wave >> 3, head = wave & 7;
  if (e >= EE) return;
  int lane = threadIdx.x & 31;
  int s = ei[e], d = ei[EE + e];
  float sc = ssrc[s * HH + head] + sdst[d * HH + head];
  float lr = sc >= 0.0f ? sc : GAT_ALPHA * sc;
  float w = expf(-lr);
  const float2* hp = (const float2*)(h + (size_t)d * DH + head * DD);
  __builtin_prefetch(hp, 0, 1);              // global_prefetch_b8
  float2 v = hp[lane];
  float* out = accum + (size_t)s * DH + head * DD + lane * 2;
  unsafeAtomicAdd(out,     w * v.x);
  unsafeAtomicAdd(out + 1, w * v.y);
}

// LayerNorm(64) + ELU per (node, head); emits f16 activations for GEMM2.
__global__ void k_ln_elu1(const float* __restrict__ acc,
                          const float* __restrict__ g,
                          const float* __restrict__ beta,
                          _Float16* __restrict__ xout) {
  int wave = (int)((blockIdx.x * blockDim.x + threadIdx.x) >> 5);
  int n = wave >> 3, head = wave & 7;
  if (n >= NN) return;
  int lane = threadIdx.x & 31;
  float2 v = ((const float2*)(acc + (size_t)n * DH + head * DD))[lane];
  float sum = v.x + v.y;
  float sq  = v.x * v.x + v.y * v.y;
#pragma unroll
  for (int m = 16; m >= 1; m >>= 1) {
    sum += __shfl_xor(sum, m, 32);
    sq  += __shfl_xor(sq,  m, 32);
  }
  float mean = sum * (1.0f / 64.0f);
  float var  = sq  * (1.0f / 64.0f) - mean * mean;
  float rs   = rsqrtf(var + GAT_LN_EPS);
  int idx = head * DD + lane * 2;
  float y0 = (v.x - mean) * rs * g[idx]     + beta[idx];
  float y1 = (v.y - mean) * rs * g[idx + 1] + beta[idx + 1];
  y0 = y0 > 0.0f ? y0 : expf(y0) - 1.0f;     // ELU (concat=True branch)
  y1 = y1 > 0.0f ? y1 : expf(y1) - 1.0f;
  xout[(size_t)n * DH + idx]     = (_Float16)y0;
  xout[(size_t)n * DH + idx + 1] = (_Float16)y1;
}

// ---------------------------------------------------------------- layer 2
__global__ void k_scores2(const float* __restrict__ h2,
                          const float* __restrict__ aout,
                          float* __restrict__ ssrc,
                          float* __restrict__ sdst) {
  int n = (int)((blockIdx.x * blockDim.x + threadIdx.x) >> 5);
  if (n >= NN) return;
  int lane = threadIdx.x & 31;
  float s0 = 0.0f, s1 = 0.0f;
#pragma unroll
  for (int r = 0; r < 4; ++r) {
    int c = lane + 32 * r;
    float2 hv = ((const float2*)(h2 + (size_t)n * FF))[c];
    float2 a0 = ((const float2*)(aout))[c];
    float2 a1 = ((const float2*)(aout + FF))[c];
    s0 += hv.x * a0.x + hv.y * a0.y;
    s1 += hv.x * a1.x + hv.y * a1.y;
  }
#pragma unroll
  for (int m = 16; m >= 1; m >>= 1) {
    s0 += __shfl_xor(s0, m, 32);
    s1 += __shfl_xor(s1, m, 32);
  }
  if (lane == 0) { ssrc[n] = s0; sdst[n] = s1; }
}

__global__ void k_scatter2(const float* __restrict__ h2,
                           const int* __restrict__ ei,
                           const float* __restrict__ ssrc,
                           const float* __restrict__ sdst,
                           float* __restrict__ accum) {
  int e = (int)((blockIdx.x * blockDim.x + threadIdx.x) >> 5);
  if (e >= EE) return;
  int lane = threadIdx.x & 31;
  int s = ei[e], d = ei[EE + e];
  float sc = ssrc[s] + sdst[d];
  float lr = sc >= 0.0f ? sc : GAT_ALPHA * sc;
  float w = expf(-lr);
  const float2* hp = (const float2*)(h2 + (size_t)d * FF);
  __builtin_prefetch(hp, 0, 1);
#pragma unroll
  for (int r = 0; r < 4; ++r) {
    int c = lane + 32 * r;
    float2 v = hp[c];
    float* out = accum + (size_t)s * FF + c * 2;
    unsafeAtomicAdd(out,     w * v.x);
    unsafeAtomicAdd(out + 1, w * v.y);
  }
}

// LayerNorm(256) (concat=False: no inner ELU) then the final elu() -> d_out.
__global__ void k_ln_elu2(const float* __restrict__ acc,
                          const float* __restrict__ g,
                          const float* __restrict__ beta,
                          float* __restrict__ out) {
  int n = (int)((blockIdx.x * blockDim.x + threadIdx.x) >> 5);
  if (n >= NN) return;
  int lane = threadIdx.x & 31;
  float2 vals[4];
  float sum = 0.0f, sq = 0.0f;
#pragma unroll
  for (int r = 0; r < 4; ++r) {
    vals[r] = ((const float2*)(acc + (size_t)n * FF))[lane + 32 * r];
    sum += vals[r].x + vals[r].y;
    sq  += vals[r].x * vals[r].x + vals[r].y * vals[r].y;
  }
#pragma unroll
  for (int m = 16; m >= 1; m >>= 1) {
    sum += __shfl_xor(sum, m, 32);
    sq  += __shfl_xor(sq,  m, 32);
  }
  float mean = sum * (1.0f / 256.0f);
  float var  = sq  * (1.0f / 256.0f) - mean * mean;
  float rs   = rsqrtf(var + GAT_LN_EPS);
#pragma unroll
  for (int r = 0; r < 4; ++r) {
    int idx = (lane + 32 * r) * 2;
    float y0 = (vals[r].x - mean) * rs * g[idx]     + beta[idx];
    float y1 = (vals[r].y - mean) * rs * g[idx + 1] + beta[idx + 1];
    y0 = y0 > 0.0f ? y0 : expf(y0) - 1.0f;   // final jax.nn.elu
    y1 = y1 > 0.0f ? y1 : expf(y1) - 1.0f;
    out[(size_t)n * FF + idx]     = y0;
    out[(size_t)n * FF + idx + 1] = y1;
  }
}

// ---------------------------------------------------------------- launch
extern "C" void kernel_launch(void* const* d_in, const int* in_sizes, int n_in,
                              void* d_out, int out_size, void* d_ws, size_t ws_size,
                              hipStream_t stream) {
  const float* embeds  = (const float*)d_in[0];   // [N,256]
  const float* W_heads = (const float*)d_in[1];   // [8,64,256] -> flat [512,256]
  const float* b_heads = (const float*)d_in[2];   // [8,64]     -> flat [512]
  const float* a_heads = (const float*)d_in[3];   // [8,128]
  const float* g_heads = (const float*)d_in[4];   // [8,64]
  const float* be_heads= (const float*)d_in[5];   // [8,64]
  const float* W_out   = (const float*)d_in[6];   // [256,512]
  const float* b_out   = (const float*)d_in[7];   // [256]
  const float* a_out   = (const float*)d_in[8];   // [512]
  const float* g_out   = (const float*)d_in[9];   // [256]
  const float* be_out  = (const float*)d_in[10];  // [256]
  const int*   ei      = (const int*)d_in[11];    // [2,E]
  float* out = (float*)d_out;

  // Workspace layout (bytes), 256-aligned, with safe aliasing.
  char* ws = (char*)d_ws;
  size_t off = 0;
  auto alloc = [&](size_t bytes) { void* p = ws + off;
                                   off += (bytes + 255) & ~(size_t)255; return p; };
  _Float16* embeds_h = (_Float16*)alloc((size_t)NN * FF * 2);   // 25.6 MB
  _Float16* W1h      = (_Float16*)alloc((size_t)DH * FF * 2);   // 0.25 MB
  _Float16* W2h      = (_Float16*)alloc((size_t)FF * DH * 2);   // 0.25 MB
  float*    h_all    = (float*)   alloc((size_t)NN * DH * 4);   // 102.4 MB
  float*    ssrc1    = (float*)   alloc((size_t)NN * HH * 4);   // 1.6 MB
  float*    sdst1    = (float*)   alloc((size_t)NN * HH * 4);   // 1.6 MB
  float*    accum1   = (float*)   alloc((size_t)NN * DH * 4);   // 102.4 MB
  _Float16* x_h      = (_Float16*)alloc((size_t)NN * DH * 2);   // 51.2 MB
  // Aliases (lifetimes do not overlap in stream order):
  float* h2     = h_all;    // [N,256] written by GEMM2 after scatter1 is done
  float* s2src  = ssrc1;    // [N]
  float* s2dst  = sdst1;    // [N]
  float* accum2 = accum1;   // [N,256] re-zeroed after LN1 consumed accum1
  (void)ws_size; (void)n_in; (void)in_sizes; (void)out_size;

  const int BLK = 256;                         // 8 wave32 per block
  // 1) f32 -> f16 conversions (A matrices + both weight matrices)
  k_f32_to_f16<<<(NN * FF + BLK - 1) / BLK, BLK, 0, stream>>>(embeds, embeds_h, NN * FF);
  k_f32_to_f16<<<(DH * FF + BLK - 1) / BLK, BLK, 0, stream>>>(W_heads, W1h, DH * FF);
  k_f32_to_f16<<<(FF * DH + BLK - 1) / BLK, BLK, 0, stream>>>(W_out,   W2h, FF * DH);

  // 2) GEMM1: h_all[N,512] = embeds @ W_heads^T + b  (all heads fused)
  {
    int tasks = (NN / 16) * (DH / 64);         // 3125 * 8 = 25000 waves
    k_gemm_wmma<<<(tasks * 32 + BLK - 1) / BLK, BLK, 0, stream>>>(
        embeds_h, W1h, b_heads, h_all, DH, FF, tasks);
  }

  // 3) attention logits per (node, head)
  k_scores1<<<(NN * HH * 32 + BLK - 1) / BLK, BLK, 0, stream>>>(h_all, a_heads, ssrc1, sdst1);

  // 4) zero + edge scatter (segment_sum via f32 atomics)
  k_zero_f32<<<(NN * DH + BLK - 1) / BLK, BLK, 0, stream>>>(accum1, NN * DH);
  k_scatter1<<<((size_t)EE * HH * 32 + BLK - 1) / BLK, BLK, 0, stream>>>(
      h_all, ei, ssrc1, sdst1, accum1);

  // 5) LayerNorm(64)+ELU -> f16 activations for layer 2
  k_ln_elu1<<<(NN * HH * 32 + BLK - 1) / BLK, BLK, 0, stream>>>(accum1, g_heads, be_heads, x_h);

  // 6) GEMM2: h2[N,256] = x @ W_out^T + b_out
  {
    int tasks = (NN / 16) * (FF / 64);         // 3125 * 4 = 12500 waves
    k_gemm_wmma<<<(tasks * 32 + BLK - 1) / BLK, BLK, 0, stream>>>(
        x_h, W2h, b_out, h2, FF, DH, tasks);
  }

  // 7) output-layer logits, zero, scatter
  k_scores2<<<(NN * 32 + BLK - 1) / BLK, BLK, 0, stream>>>(h2, a_out, s2src, s2dst);
  k_zero_f32<<<(NN * FF + BLK - 1) / BLK, BLK, 0, stream>>>(accum2, NN * FF);
  k_scatter2<<<((size_t)EE * 32 + BLK - 1) / BLK, BLK, 0, stream>>>(
      h2, ei, s2src, s2dst, accum2);

  // 8) LayerNorm(256) + final ELU -> d_out
  k_ln_elu2<<<(NN * 32 + BLK - 1) / BLK, BLK, 0, stream>>>(accum2, g_out, be_out, out);
}
